// AttentionHead_7859790151895
// MI455X (gfx1250) — matrix operand order repaired and checked by reading, not compiled, not run on hardware
//
#include <hip/hip_runtime.h>

// AttentionHead for MI455X (gfx1250, wave32, WMMA).
// Kernel 1: fused QKV projection (f32 -> f16, WMMA f32_16x16x32_f16, scale folded into Q).
// Kernel 2: flash attention, 16 query rows per wave, 32-key blocks, online softmax.
//           V blocks are prefetched memory->LDS with double buffering via the
//           CDNA5 async path (global_load_async_to_lds_b128 + s_wait_asynccnt).
// Workspace usage: 3 * B*T*H f16 = 6 MB.

#define B_ 4
#define T_ 4096
#define E_ 1024
#define H_ 64

typedef __attribute__((ext_vector_type(16))) _Float16 v16h;
typedef __attribute__((ext_vector_type(8)))  _Float16 v8h;
typedef __attribute__((ext_vector_type(8)))  float    v8f;

__device__ __forceinline__ v8f wmma_f16(v16h a, v16h b, v8f c) {
  // (neg_a, A, neg_b, B, c_mod, C, reuse_a, reuse_b)
  return __builtin_amdgcn_wmma_f32_16x16x32_f16(false, a, false, b, (short)0, c,
                                                false, false);
}

// Async copy of one contiguous 4 KB block (32 key rows x 128 B) from global
// memory directly into LDS.  Each of the 32 lanes owns a 16 B slice; the
// instruction offset applies to BOTH the global and LDS address (ISA 10.x):
__device__ __forceinline__ void async_copy_4k(uint32_t lds_base,
                                              const char* gbase, int lane) {
  uint32_t l = lds_base + lane * 16;
  const char* g = gbase + lane * 16;
  asm volatile("global_load_async_to_lds_b128 %0, %1, off"             :: "v"(l), "v"(g) : "memory");
  asm volatile("global_load_async_to_lds_b128 %0, %1, off offset:512"  :: "v"(l), "v"(g) : "memory");
  asm volatile("global_load_async_to_lds_b128 %0, %1, off offset:1024" :: "v"(l), "v"(g) : "memory");
  asm volatile("global_load_async_to_lds_b128 %0, %1, off offset:1536" :: "v"(l), "v"(g) : "memory");
  asm volatile("global_load_async_to_lds_b128 %0, %1, off offset:2048" :: "v"(l), "v"(g) : "memory");
  asm volatile("global_load_async_to_lds_b128 %0, %1, off offset:2560" :: "v"(l), "v"(g) : "memory");
  asm volatile("global_load_async_to_lds_b128 %0, %1, off offset:3072" :: "v"(l), "v"(g) : "memory");
  asm volatile("global_load_async_to_lds_b128 %0, %1, off offset:3584" :: "v"(l), "v"(g) : "memory");
}

// ---------------------------------------------------------------------------
// Kernel 1: Q/K/V projection.  One wave per 16-row tile of (b*T+t).
// A = emb tile [16 x 32] (f32 -> f16), B = W^T tile [32 x 16] (W is [H,E] row
// major, so B[k=e][n=h] = W[h,e] is contiguous in e per lane -> vectorizable).
// ---------------------------------------------------------------------------
__global__ __launch_bounds__(32) void qkv_proj(
    const float* __restrict__ emb,
    const float* __restrict__ Wq, const float* __restrict__ Wk,
    const float* __restrict__ Wv,
    _Float16* __restrict__ Q, _Float16* __restrict__ K, _Float16* __restrict__ V) {
  const int lane  = threadIdx.x & 31;
  const long row0 = (long)blockIdx.x * 16;      // flattened b*T + t
  const int n     = lane & 15;                  // A: row M ; B: col N ; C: col N
  const int aklo  = (lane & 16) ? 16 : 0;       // A-layout K split per lane half
  const int bk    = (lane & 16) ? 16 : 0;       // B-layout K split per lane half

  v8f accq[4] = {}, acck[4] = {}, accv[4] = {};

  const float* arow = emb + (row0 + n) * E_;
  for (int e0 = 0; e0 < E_; e0 += 32) {
    v16h a;
#pragma unroll
    for (int i = 0; i < 8; ++i) {
      a[i]     = (_Float16)arow[e0 + aklo + i];
      a[8 + i] = (_Float16)arow[e0 + aklo + 8 + i];
    }
#pragma unroll
    for (int j = 0; j < 4; ++j) {
      const int h = j * 16 + n;
      const float* wq = Wq + (long)h * E_ + e0 + bk;
      const float* wk = Wk + (long)h * E_ + e0 + bk;
      const float* wv = Wv + (long)h * E_ + e0 + bk;
      v16h bq, bkm, bv;
#pragma unroll
      for (int i = 0; i < 16; ++i) {
        bq[i]  = (_Float16)wq[i];
        bkm[i] = (_Float16)wk[i];
        bv[i]  = (_Float16)wv[i];
      }
      accq[j] = wmma_f16(a, bq,  accq[j]);
      acck[j] = wmma_f16(a, bkm, acck[j]);
      accv[j] = wmma_f16(a, bv,  accv[j]);
    }
  }

  const int mrow = (lane & 16) ? 8 : 0;         // C-layout: VGPR r -> row r+8*hi
  const float qscale = 0.03125f;                // E^-0.5 = 1/32 folded into Q
#pragma unroll
  for (int j = 0; j < 4; ++j) {
#pragma unroll
    for (int r = 0; r < 8; ++r) {
      const long orow = row0 + r + mrow;
      const int  h    = j * 16 + n;
      Q[orow * H_ + h] = (_Float16)(accq[j][r] * qscale);
      K[orow * H_ + h] = (_Float16)(acck[j][r]);
      V[orow * H_ + h] = (_Float16)(accv[j][r]);
    }
  }
}

// ---------------------------------------------------------------------------
// Kernel 2: flash attention.  grid = (T/16, B), one wave per block.
// ---------------------------------------------------------------------------
__global__ __launch_bounds__(32) void flash_attn(
    const _Float16* __restrict__ Q, const _Float16* __restrict__ K,
    const _Float16* __restrict__ V, float* __restrict__ out) {
  __shared__ _Float16 lds_p[16 * 32];        // P tile (16 q x 32 keys), row major
  __shared__ _Float16 lds_v[2][32 * H_];     // double-buffered V block (4 KB each)

  const int lane = threadIdx.x & 31;
  const int b    = blockIdx.y;
  const int q0   = blockIdx.x * 16;

  const _Float16* Qb = Q + ((long)b * T_ + q0) * H_;
  const _Float16* Kb = K + (long)b * T_ * H_;
  const _Float16* Vb = V + (long)b * T_ * H_;

  const int n     = lane & 15;
  const int hi8   = (lane & 16) ? 8 : 0;
  const int aklo  = (lane & 16) ? 16 : 0;
  const int bkofs = (lane & 16) ? 16 : 0;

  // Q as two A tiles (K-dim = head dim 0..31 / 32..63); 16B-aligned vec loads.
  v16h qa0, qa1;
  {
    const _Float16* row = Qb + n * H_;
    v8h l0 = *(const v8h*)(row + aklo);
    v8h h0 = *(const v8h*)(row + aklo + 8);
    v8h l1 = *(const v8h*)(row + 32 + aklo);
    v8h h1 = *(const v8h*)(row + 32 + aklo + 8);
#pragma unroll
    for (int i = 0; i < 8; ++i) {
      qa0[i] = l0[i]; qa0[8 + i] = h0[i];
      qa1[i] = l1[i]; qa1[8 + i] = h1[i];
    }
  }

  v8f o0 = {}, o1 = {}, o2 = {}, o3 = {};
  float mrow[8], lrow[8];
#pragma unroll
  for (int r = 0; r < 8; ++r) { mrow[r] = -1e30f; lrow[r] = 0.0f; }

  const int kend = q0 + 16;                     // keys < q0+16 (causal)

  // Prefetch first V block (rows 0..31 are a contiguous 4 KB chunk of V).
  async_copy_4k((uint32_t)(uintptr_t)&lds_v[0][0], (const char*)Vb, lane);

  int buf = 0;
  for (int k0 = 0; k0 < kend; k0 += 32) {
    const bool has_next = (k0 + 32) < kend;
    if (has_next)  // prefetch next V block into the other buffer
      async_copy_4k((uint32_t)(uintptr_t)&lds_v[buf ^ 1][0],
                    (const char*)(Vb + (long)(k0 + 32) * H_), lane);

    // ---- scores: S[16q x 32k] = Q * K^T  (B lane = key, halves = head dim)
    v8f s0 = {}, s1 = {};
    {
      const _Float16* kr0 = Kb + (long)(k0 + n) * H_;
      const _Float16* kr1 = Kb + (long)(k0 + 16 + n) * H_;
      v16h b00, b01, b10, b11;
#pragma unroll
      for (int i = 0; i < 16; ++i) {
        b00[i] = kr0[bkofs + i];
        b01[i] = kr0[32 + bkofs + i];
        b10[i] = kr1[bkofs + i];
        b11[i] = kr1[32 + bkofs + i];
      }
      s0 = wmma_f16(qa0, b00, s0);
      s0 = wmma_f16(qa1, b01, s0);
      s1 = wmma_f16(qa0, b10, s1);
      s1 = wmma_f16(qa1, b11, s1);
    }

    // ---- causal mask on diagonal blocks
    if (k0 + 31 > q0) {
#pragma unroll
      for (int r = 0; r < 8; ++r) {
        const int qidx = q0 + r + hi8;
        if (k0 + n > qidx)      s0[r] = -1e30f;
        if (k0 + 16 + n > qidx) s1[r] = -1e30f;
      }
    }

    // ---- wait for the CURRENT V buffer; async loads complete in order, so
    // allowing 8 outstanding leaves only this iteration's prefetch in flight.
    if (has_next) asm volatile("s_wait_asynccnt 0x8" ::: "memory");
    else          asm volatile("s_wait_asynccnt 0x0" ::: "memory");

    // ---- online softmax (per query row; row stats replicated per lane half)
#pragma unroll
    for (int r = 0; r < 8; ++r) {
      float vmax = fmaxf(s0[r], s1[r]);
      vmax = fmaxf(vmax, __shfl_xor(vmax, 1, 32));
      vmax = fmaxf(vmax, __shfl_xor(vmax, 2, 32));
      vmax = fmaxf(vmax, __shfl_xor(vmax, 4, 32));
      vmax = fmaxf(vmax, __shfl_xor(vmax, 8, 32));
      const float mnew = fmaxf(mrow[r], vmax);
      const float corr = __expf(mrow[r] - mnew);
      mrow[r] = mnew;
      const float p0 = __expf(s0[r] - mnew);
      const float p1 = __expf(s1[r] - mnew);
      float psum = p0 + p1;
      psum += __shfl_xor(psum, 1, 32);
      psum += __shfl_xor(psum, 2, 32);
      psum += __shfl_xor(psum, 4, 32);
      psum += __shfl_xor(psum, 8, 32);
      lrow[r] = lrow[r] * corr + psum;
      o0[r] *= corr; o1[r] *= corr; o2[r] *= corr; o3[r] *= corr;
      lds_p[(r + hi8) * 32 + n]      = (_Float16)p0;   // C-layout -> row major
      lds_p[(r + hi8) * 32 + 16 + n] = (_Float16)p1;
    }

    // ---- P as A tile (16 x 32 keys) from LDS
    v16h pa;
    {
      const _Float16* prow = lds_p + n * 32;
#pragma unroll
      for (int i = 0; i < 8; ++i) {
        pa[i]     = prow[aklo + i];
        pa[8 + i] = prow[aklo + 8 + i];
      }
    }

    // ---- O += P * V  (B lane = h, halves = key; strided LDS reads)
    const _Float16* vbuf = &lds_v[buf][0];
#pragma unroll
    for (int j = 0; j < 4; ++j) {
      v16h vb;
#pragma unroll
      for (int i = 0; i < 16; ++i)
        vb[i] = vbuf[(bkofs + i) * H_ + j * 16 + n];
      if (j == 0)      o0 = wmma_f16(pa, vb, o0);
      else if (j == 1) o1 = wmma_f16(pa, vb, o1);
      else if (j == 2) o2 = wmma_f16(pa, vb, o2);
      else             o3 = wmma_f16(pa, vb, o3);
    }
    buf ^= 1;
    // single wave per workgroup: DS ops in-order within the wave, no barrier
  }

  // ---- normalize and write fp32 output
#pragma unroll
  for (int r = 0; r < 8; ++r) {
    const float inv = 1.0f / lrow[r];
    const long row = (long)b * T_ + q0 + r + hi8;
    out[row * H_ +  0 + n] = o0[r] * inv;
    out[row * H_ + 16 + n] = o1[r] * inv;
    out[row * H_ + 32 + n] = o2[r] * inv;
    out[row * H_ + 48 + n] = o3[r] * inv;
  }
}

extern "C" void kernel_launch(void* const* d_in, const int* in_sizes, int n_in,
                              void* d_out, int out_size, void* d_ws, size_t ws_size,
                              hipStream_t stream) {
  const float* emb = (const float*)d_in[0];
  const float* Wq  = (const float*)d_in[1];
  const float* Wk  = (const float*)d_in[2];
  const float* Wv  = (const float*)d_in[3];

  const size_t qkv_elems = (size_t)B_ * T_ * H_;   // 1M f16 elems each
  _Float16* Q = (_Float16*)d_ws;
  _Float16* K = Q + qkv_elems;
  _Float16* V = K + qkv_elems;

  qkv_proj<<<dim3((B_ * T_) / 16), dim3(32), 0, stream>>>(emb, Wq, Wk, Wv, Q, K, V);
  flash_attn<<<dim3(T_ / 16, B_), dim3(32), 0, stream>>>(Q, K, V, (float*)d_out);
}
